// PVCNNSymLearnedPlane_17446157156941
// MI455X (gfx1250) — matrix operand order repaired and checked by reading, not compiled, not run on hardware
//
#include <hip/hip_runtime.h>
#include <hip/hip_bf16.h>
#include <math.h>

typedef __attribute__((ext_vector_type(16))) _Float16 v16h;
typedef __attribute__((ext_vector_type(8)))  _Float16 v8h;
typedef __attribute__((ext_vector_type(8)))  float    v8f;

#define BATCH 32
#define NPTS  32768
#define HID   64
#define PPB   128   // points per block (8 waves x 16)
#define ROWS  72    // padded LDS row stride in halves (144B: 16B-aligned, bank-conflict-free)

__device__ __forceinline__ float silu_f(float v) {
    // v * sigmoid(v); fast rcp instead of IEEE divide (epilogue of f16 WMMA,
    // feeding a max-pool -> 1-ulp rcp error is irrelevant)
    return v * __builtin_amdgcn_rcpf(1.0f + __expf(-v));
}

__device__ __forceinline__ void atomicMaxFloat(float* addr, float val) {
    // sign-aware integer atomic max on float bit patterns (init must be -inf)
    if (val >= 0.0f) atomicMax((int*)addr, __float_as_int(val));
    else             atomicMin((unsigned int*)addr, __float_as_uint(val));
}

__device__ __forceinline__ v16h cat16(v8h lo, v8h hi) {
    return __builtin_shufflevector(lo, hi, 0, 1, 2, 3, 4, 5, 6, 7,
                                           8, 9, 10, 11, 12, 13, 14, 15);
}

// ---------------------------------------------------------------- init
__global__ void init_feat_kernel(float* gfeat, int n) {
    int i = blockIdx.x * blockDim.x + threadIdx.x;
    if (i < n) gfeat[i] = -INFINITY;
}

// ---------------------------------------------------------------- pointnet (WMMA)
// feats = silu(silu(x@W1+b1)@W2+b2); gfeat[b][h] = max over points (atomic)
__global__ __launch_bounds__(256, 2)
void pointnet_kernel(const float* __restrict__ x_t,
                     const float* __restrict__ W1, const float* __restrict__ b1,
                     const float* __restrict__ W2, const float* __restrict__ b2,
                     float* __restrict__ gfeat) {
    __shared__ __align__(16) _Float16 sW2T[HID * ROWS]; // W2 transposed: [n][k], f16
    __shared__ __align__(16) _Float16 sH1[PPB * ROWS];  // layer-1 activations [p][k], f16
    __shared__ float sW1[3 * HID];
    __shared__ float sB1[HID];
    __shared__ float sB2[HID];
    __shared__ float sMax[8][HID];                      // per-wave column maxima

    const int t    = threadIdx.x;
    const int b    = blockIdx.y;
    const int base = blockIdx.x * PPB;

    for (int i = t; i < HID * HID; i += 256) {
        const int k = i >> 6, n = i & 63;
        sW2T[n * ROWS + k] = (_Float16)W2[i];           // transpose: contiguous k per n
    }
    if (t < 3 * HID) sW1[t] = W1[t];
    if (t < HID) { sB1[t] = b1[t]; sB2[t] = b2[t]; }
    __syncthreads();

    // ---- layer 1 (K=3, fp32 VALU): 2 threads per point, 32 dims each
    {
        const int p  = t >> 1;
        const int j0 = (t & 1) * 32;
        const float* xp = x_t + ((long long)b * NPTS + base + p) * 3;
        const float x0 = xp[0], x1 = xp[1], x2 = xp[2];
#pragma unroll 8
        for (int j = j0; j < j0 + 32; ++j) {
            float v = fmaf(x0, sW1[j],
                      fmaf(x1, sW1[HID + j],
                      fmaf(x2, sW1[2 * HID + j], sB1[j])));
            sH1[p * ROWS + j] = (_Float16)silu_f(v);
        }
    }
    __syncthreads();

    // ---- layer 2 (K=64): v_wmma_f32_16x16x32_f16, 16 points per wave
    const int wave  = t >> 5;
    const int lane  = t & 31;
    const int mbase = wave * 16;
    const int mlane = lane & 15;
    const int hiA   = (lane >= 16) ? 8 : 0;    // ISA 16-bit A 16x32 layout
    const int hiB   = (lane >= 16) ? 16 : 0;   // ISA 16-bit B 32x16 layout

    // A fragments: two contiguous 16B runs each -> ds_load_b128 pairs
    const v8h* rowA = (const v8h*)(sH1 + (mbase + mlane) * ROWS);
    const int  ca   = hiA >> 3;                // chunk = 8 halves
    const v16h a0 = cat16(rowA[ca],     rowA[ca + 2]);
    const v16h a1 = cat16(rowA[ca + 4], rowA[ca + 6]);

#pragma unroll
    for (int nt = 0; nt < 4; ++nt) {
        const int n = mlane + nt * 16;
        // B fragments: contiguous k runs in the transposed W2 row for column n
        const v8h* rowB = (const v8h*)(sW2T + n * ROWS);
        const int  cb   = hiB >> 4;
        const v16h bf0 = cat16(rowB[cb],     rowB[cb + 2]);
        const v16h bf1 = cat16(rowB[cb + 4], rowB[cb + 6]);

        v8f c = {};
        c = __builtin_amdgcn_wmma_f32_16x16x32_f16(false, a0, false, bf0,
                                                   (short)0, c, false, false);
        c = __builtin_amdgcn_wmma_f32_16x16x32_f16(false, a1, false, bf1,
                                                   (short)0, c, false, false);
        const float bias = sB2[n];
        float vmax = -INFINITY;
#pragma unroll
        for (int r = 0; r < 8; ++r) {
            vmax = fmaxf(vmax, silu_f(c[r] + bias));
        }
        vmax = fmaxf(vmax, __shfl_xor(vmax, 16, 32));  // merge M halves (wave32)
        if (lane < 16) sMax[wave][n] = vmax;
    }
    __syncthreads();

    // block-level reduction, then one atomic per column
    if (t < HID) {
        float m = sMax[0][t];
#pragma unroll
        for (int w = 1; w < 8; ++w) m = fmaxf(m, sMax[w][t]);
        atomicMaxFloat(gfeat + b * HID + t, m);
    }
}

// ---------------------------------------------------------------- head MLP
// n = normalize(silu(cat[gfeat, time_embed(t)] @ Wh1 + bh1) @ Wh2 + bh2)
__global__ __launch_bounds__(64)
void head_kernel(const int* __restrict__ tarr, const float* __restrict__ gfeat,
                 const float* __restrict__ Wh1, const float* __restrict__ bh1,
                 const float* __restrict__ Wh2, const float* __restrict__ bh2,
                 float* __restrict__ nbuf, float* __restrict__ outN) {
    __shared__ float in128[128];
    __shared__ float hid[HID];
    const int b = blockIdx.x;
    const int j = threadIdx.x;   // 0..63
    const float tv = (float)tarr[b];

    in128[j] = gfeat[b * HID + j];
    {
        const int i = j & 31;
        const float freq = __expf(-logf(10000.0f) * (float)i / 32.0f);
        const float arg = tv * freq;
        in128[64 + j] = (j < 32) ? sinf(arg) : cosf(arg);
    }
    __syncthreads();

    float acc = bh1[j];
#pragma unroll 4
    for (int i = 0; i < 128; ++i) acc = fmaf(in128[i], Wh1[i * HID + j], acc);
    hid[j] = acc * __builtin_amdgcn_rcpf(1.0f + __expf(-acc));
    __syncthreads();

    if (j == 0) {
        float o[3];
#pragma unroll
        for (int c = 0; c < 3; ++c) {
            float a = bh2[c];
            for (int i = 0; i < HID; ++i) a = fmaf(hid[i], Wh2[i * 3 + c], a);
            o[c] = a;
        }
        float nrm = sqrtf(o[0] * o[0] + o[1] * o[1] + o[2] * o[2]);
        nrm = fmaxf(nrm, 1e-6f);
#pragma unroll
        for (int c = 0; c < 3; ++c) {
            const float v = o[c] / nrm;
            nbuf[b * 3 + c] = v;
            outN[b * 3 + c] = v;
        }
    }
}

// ---------------------------------------------------------------- topk via full
// in-LDS bitonic sort (256 KB keys -> only possible with CDNA5's 320 KB LDS)
__global__ __launch_bounds__(1024, 1)
void select_kernel(const float* __restrict__ x_t, const float* __restrict__ nbuf,
                   float* __restrict__ outX, int* __restrict__ outIdx) {
    __shared__ unsigned long long keys[NPTS];   // 256 KB
    const int b = blockIdx.x;
    const int K = NPTS / 2;
    const float n0 = nbuf[b * 3 + 0];
    const float n1 = nbuf[b * 3 + 1];
    const float n2 = nbuf[b * 3 + 2];
    const float* xb = x_t + (long long)b * NPTS * 3;

    // scores monotone in proj (TAU>0, uniform shift) -> sort by proj directly.
    for (int i = threadIdx.x; i < NPTS; i += blockDim.x) {
        const float p = xb[i * 3] * n0 + xb[i * 3 + 1] * n1 + xb[i * 3 + 2] * n2;
        unsigned u = __float_as_uint(p);
        unsigned s = (u & 0x80000000u) ? ~u : (u | 0x80000000u);  // order-preserving
        keys[i] = ((unsigned long long)s << 32) | (unsigned)(~i); // tie: low idx first
    }

    // bitonic sort, descending
    for (int k = 2; k <= NPTS; k <<= 1) {
        for (int j = k >> 1; j > 0; j >>= 1) {
            __syncthreads();
            for (int i = threadIdx.x; i < NPTS; i += blockDim.x) {
                const int ixj = i ^ j;
                if (ixj > i) {
                    const unsigned long long A = keys[i];
                    const unsigned long long Bk = keys[ixj];
                    const bool up = ((i & k) == 0);
                    if (up ? (A < Bk) : (A > Bk)) { keys[i] = Bk; keys[ixj] = A; }
                }
            }
        }
    }
    __syncthreads();

    for (int r = threadIdx.x; r < K; r += blockDim.x) {
        const unsigned long long key = keys[r];
        const int i = (int)(~(unsigned)key);
        outIdx[b * K + r] = i;
        const long long o = ((long long)b * K + r) * 3;
        outX[o + 0] = xb[i * 3 + 0];
        outX[o + 1] = xb[i * 3 + 1];
        outX[o + 2] = xb[i * 3 + 2];
    }
}

// ---------------------------------------------------------------- launch
extern "C" void kernel_launch(void* const* d_in, const int* in_sizes, int n_in,
                              void* d_out, int out_size, void* d_ws, size_t ws_size,
                              hipStream_t stream) {
    const float* x_t = (const float*)d_in[0];
    const int*   t   = (const int*)  d_in[1];
    const float* W1  = (const float*)d_in[2];
    const float* b1  = (const float*)d_in[3];
    const float* W2  = (const float*)d_in[4];
    const float* b2  = (const float*)d_in[5];
    const float* Wh1 = (const float*)d_in[6];
    const float* bh1 = (const float*)d_in[7];
    const float* Wh2 = (const float*)d_in[8];
    const float* bh2 = (const float*)d_in[9];

    const int K = NPTS / 2;
    float* gfeat = (float*)d_ws;                 // [32][64]
    float* nbuf  = gfeat + BATCH * HID;          // [32][3]

    // output layout: X_half [32][K][3] f32, indices [32][K] i32, n [32][3] f32
    float* outX   = (float*)d_out;
    int*   outIdx = (int*)((float*)d_out + (size_t)BATCH * K * 3);
    float* outN   = (float*)d_out + (size_t)BATCH * K * 3 + (size_t)BATCH * K;

    init_feat_kernel<<<(BATCH * HID + 255) / 256, 256, 0, stream>>>(gfeat, BATCH * HID);

    pointnet_kernel<<<dim3(NPTS / PPB, BATCH), 256, 0, stream>>>(
        x_t, W1, b1, W2, b2, gfeat);

    head_kernel<<<BATCH, 64, 0, stream>>>(t, gfeat, Wh1, bh1, Wh2, bh2, nbuf, outN);

    select_kernel<<<BATCH, 1024, 0, stream>>>(x_t, nbuf, outX, outIdx);
}
